// BaseHORN_24309514895360
// MI455X (gfx1250) — matrix operand order repaired and checked by reading, not compiled, not run on hardware
//
#include <hip/hip_runtime.h>

typedef __attribute__((ext_vector_type(16))) __bf16 v16bf;
typedef __attribute__((ext_vector_type(8)))  float  v8f;

#define T_STEPS 784
#define NN      256
#define TILE_B  16
#define YSTRIDE 274   // shorts = 137 dwords; makes A-fragment lane->bank map conflict-free

struct U8x { unsigned int u[8]; };

__device__ __forceinline__ unsigned short f2bf(float f) {
  unsigned int u = __float_as_uint(f);
  unsigned int r = u + 0x7FFFu + ((u >> 16) & 1u);   // round-to-nearest-even
  return (unsigned short)(r >> 16);
}

__device__ __forceinline__ float fast_tanh(float x) {
#if __has_builtin(__builtin_amdgcn_tanh_f32)
  return __builtin_amdgcn_tanh_f32(x);
#else
  return tanhf(x);
#endif
}

__global__ __launch_bounds__(256, 1)
void horn_scan_kernel(const float* __restrict__ seq,    // [B][784]
                      const float* __restrict__ wih_w,  // [256]
                      const float* __restrict__ wih_b,  // [256]
                      const float* __restrict__ whh,    // [256][256]
                      const float* __restrict__ whh_b,  // [256]
                      const float* __restrict__ ro_w,   // [10][256]
                      const float* __restrict__ ro_b,   // [10]
                      float* __restrict__ out)          // [B][10]
{
  __shared__ unsigned short ylds[2][TILE_B * YSTRIDE]; // ping-pong bf16 y (A operand)
  __shared__ float seq_lds[2][TILE_B];                 // ping-pong staged seq[b,t]
  __shared__ float xsh[TILE_B][NN];                    // final x for readout

  const int tid  = threadIdx.x;
  const int wave = tid >> 5;
  const int lane = tid & 31;
  const int m    = lane & 15;   // N-index within tile (= recv node col / batch col)
  const int hi   = lane >> 4;
  const int bbase = blockIdx.x * TILE_B;

  // ---- one-time: W_hh^T -> bf16 B-operand fragments, weight-stationary in VGPRs ----
  // B layout (32x16 bf16): VGPR v holds K = {2v,2v+1} (v<4) / {16+2(v-4),..} (v>=4) for lanes 0-15; lanes 16-31: K+8, N = lane%16
  unsigned int bfr[2][8][8];
  float wihw_r[2], wihb_r[2], whhb_r[2];
  #pragma unroll
  for (int nt = 0; nt < 2; ++nt) {
    const int n = (wave * 2 + nt) * 16 + m;
    wihw_r[nt] = wih_w[n];
    wihb_r[nt] = wih_b[n];
    whhb_r[nt] = whh_b[n];
    const float* wr = whh + n * NN;                    // Wt[k][n] = W_hh[n][k]
    #pragma unroll
    for (int ks = 0; ks < 8; ++ks) {
      #pragma unroll
      for (int v = 0; v < 8; ++v) {
        const int k = ks * 32 + ((v < 4) ? (2 * v) : (16 + 2 * (v - 4))) + 8 * hi;
        bfr[nt][ks][v] = (unsigned)f2bf(wr[k]) | ((unsigned)f2bf(wr[k + 1]) << 16);
      }
    }
  }

  // init: y(bf16) buffer 0 = 0, stage seq for t=0
  for (int i = tid; i < TILE_B * YSTRIDE; i += 256) ylds[0][i] = 0;
  if (tid < TILE_B) seq_lds[0][tid] = seq[(bbase + tid) * T_STEPS + 0];
  __syncthreads();

  float xs[2][8], ys[2][8];
  #pragma unroll
  for (int nt = 0; nt < 2; ++nt)
    #pragma unroll
    for (int r = 0; r < 8; ++r) { xs[nt][r] = 0.f; ys[nt][r] = 0.f; }

  const float scale = 0.0625f;   // 1/sqrt(256)

  for (int t = 0; t < T_STEPS; ++t) {
    const int pp = t & 1;
    // prefetch next step's input scalars into the other ping-pong slot
    if (tid < TILE_B && (t + 1) < T_STEPS)
      seq_lds[pp ^ 1][tid] = seq[(bbase + tid) * T_STEPS + (t + 1)];

    v8f acc[2] = {v8f{}, v8f{}};

    // I_rec tile = y[16 x 256] @ Whh^T[256 x 16]: 8 k-slices of bf16 WMMA
    #pragma unroll
    for (int ks = 0; ks < 8; ++ks) {
      // A fragment (16x32 bf16), shared by both n-tiles of this wave.
      // A layout: lane<16: M=lane, K = {2v,2v+1} (v<4) / {16+2(v-4),..} (v>=4); lanes>=16: K+8
      U8x au;
      const unsigned short* yrow = &ylds[pp][m * YSTRIDE + ks * 32 + 8 * hi];
      #pragma unroll
      for (int v = 0; v < 8; ++v) {
        const int off = (v < 4) ? (2 * v) : (16 + 2 * (v - 4));
        au.u[v] = *(const unsigned int*)(yrow + off);
      }
      const v16bf av = __builtin_bit_cast(v16bf, au);
      #pragma unroll
      for (int nt = 0; nt < 2; ++nt) {
        U8x bu;
        #pragma unroll
        for (int v = 0; v < 8; ++v) bu.u[v] = bfr[nt][ks][v];
        const v16bf bv = __builtin_bit_cast(v16bf, bu);
        acc[nt] = __builtin_amdgcn_wmma_f32_16x16x32_bf16(
            false, av, false, bv, (short)0, acc[nt], false, false);
      }
    }

    // elementwise oscillator update (fp32 state in registers), publish y as bf16
    #pragma unroll
    for (int nt = 0; nt < 2; ++nt) {
      const int n = (wave * 2 + nt) * 16 + m;
      #pragma unroll
      for (int r = 0; r < 8; ++r) {
        const float irec = acc[nt][r] + whhb_r[nt];
        const float iext = seq_lds[pp][8 * hi + r] * wihw_r[nt] + wihb_r[nt];
        const float A    = (iext + irec) * scale;
        float x = xs[nt][r], y = ys[nt][r];
        const float accel = fast_tanh(A) - 0.2f * y - x;
        x += y;
        y += accel;
        xs[nt][r] = x; ys[nt][r] = y;
        ylds[pp ^ 1][(r + 8 * hi) * YSTRIDE + n] = f2bf(y);
      }
    }
    __syncthreads();   // single barrier per step (ping-pong buffers)
  }

  // ---- readout: out = x_final @ ro_w^T + ro_b ----
  #pragma unroll
  for (int nt = 0; nt < 2; ++nt)
    #pragma unroll
    for (int r = 0; r < 8; ++r)
      xsh[r + 8 * hi][(wave * 2 + nt) * 16 + m] = xs[nt][r];
  __syncthreads();

  if (tid < TILE_B * 10) {
    const int b = tid / 10, c = tid % 10;
    float a = ro_b[c];
    const float* rw = ro_w + c * NN;
    #pragma unroll 8
    for (int n = 0; n < NN; ++n) a += xsh[b][n] * rw[n];
    out[(bbase + b) * 10 + c] = a;
  }
}

extern "C" void kernel_launch(void* const* d_in, const int* in_sizes, int n_in,
                              void* d_out, int out_size, void* d_ws, size_t ws_size,
                              hipStream_t stream) {
  const float* seq    = (const float*)d_in[0];
  const float* wih_w  = (const float*)d_in[1];
  const float* wih_b  = (const float*)d_in[2];
  const float* whh    = (const float*)d_in[3];
  const float* whh_b  = (const float*)d_in[4];
  const float* ro_w   = (const float*)d_in[5];
  const float* ro_b   = (const float*)d_in[6];
  float* out = (float*)d_out;

  const int B = in_sizes[0] / T_STEPS;       // 1024
  const int grid = B / TILE_B;               // 64 workgroups, 8 wave32 each
  horn_scan_kernel<<<grid, 256, 0, stream>>>(seq, wih_w, wih_b, whh, whh_b,
                                             ro_w, ro_b, out);
}
